// GNNModule_4148938408095
// MI455X (gfx1250) — matrix operand order repaired and checked by reading, not compiled, not run on hardware
//
#include <hip/hip_runtime.h>
#include <hip/hip_bf16.h>

#define NN 50000
#define EE 800000
#define ND 128
#define ED 16
#define HH 64
#define LL 3
#define GG 64

typedef __bf16 bf16_t;
typedef __attribute__((ext_vector_type(16))) __bf16 v16bf;
typedef __attribute__((ext_vector_type(8)))  float  v8f;

static __device__ __forceinline__ v8f wmma_bf16(v16bf a, v16bf b, v8f c) {
  return __builtin_amdgcn_wmma_f32_16x16x32_bf16(false, a, false, b, (short)0, c, false, false);
}

static __device__ __forceinline__ v8f zero8() {
  v8f z;
#pragma unroll
  for (int i = 0; i < 8; ++i) z[i] = 0.0f;
  return z;
}

// ---- gfx1250 async global->LDS (per-lane 16B move, tracked by ASYNCcnt) ----
static __device__ __forceinline__ unsigned lds_off(const void* p) {
  // LDS aperture: flat addr[31:0] is the LDS byte address (ISA 10.2).
  return (unsigned)(size_t)p;
}
static __device__ __forceinline__ void async_b128(unsigned lds_byte, const void* g) {
  asm volatile("global_load_async_to_lds_b128 %0, %1, off"
               :: "v"(lds_byte), "v"((unsigned long long)(size_t)g)
               : "memory");
}
static __device__ __forceinline__ void async_wait0() {
  asm volatile("s_wait_asynccnt 0" ::: "memory");
}

// A fragment (16x32 bf16) from LDS row-major tile; stride in elements.
static __device__ __forceinline__ v16bf ld_afrag(const bf16_t* lds, int row0, int stride,
                                                 int k0, int lane) {
  const int m  = lane & 15;
  const int hk = (lane >> 4) << 4;
  const bf16_t* p = lds + (row0 + m) * stride + k0 + hk;
  v16bf a;
#pragma unroll
  for (int e = 0; e < 16; ++e) a[e] = p[e];
  return a;
}

// B fragment (32x16 bf16): LDS holds B transposed [N][K].
static __device__ __forceinline__ v16bf ld_bfragT(const bf16_t* ldsT, int n0, int strideK,
                                                  int k0, int lane) {
  const int n  = lane & 15;
  const int kb = (lane >> 4) << 4;
  const bf16_t* p = ldsT + (n0 + n) * strideK + k0 + kb;
  v16bf b;
#pragma unroll
  for (int e = 0; e < 16; ++e) b[e] = p[e];
  return b;
}

// B fragment from 16 contiguous f32 in global (convert on the fly).
static __device__ __forceinline__ v16bf ld_bfrag_g(const float* p) {
  v16bf b;
#pragma unroll
  for (int e = 0; e < 16; ++e) b[e] = (bf16_t)p[e];
  return b;
}

// ------------- prep: bf16 images of edge_attr and transposed MLP weights -------------
__global__ __launch_bounds__(256)
void k_prep_attr(const float* __restrict__ src, bf16_t* __restrict__ dst) {
  int i4 = blockIdx.x * 256 + threadIdx.x;
  if (i4 >= EE * ED / 4) return;
  float4 v = *(const float4*)(src + (size_t)i4 * 4);
  bf16_t* d = dst + (size_t)i4 * 4;
  d[0] = (bf16_t)v.x; d[1] = (bf16_t)v.y; d[2] = (bf16_t)v.z; d[3] = (bf16_t)v.w;
}

__global__ __launch_bounds__(256)
void k_prep_w(const float* __restrict__ msg_w1, const float* __restrict__ msg_w2,
              bf16_t* __restrict__ w1T, bf16_t* __restrict__ w2T) {
  const int l = blockIdx.x, tid = threadIdx.x;
  const float* w1 = msg_w1 + (size_t)l * (2 * HH + ED) * HH;
  const float* w2 = msg_w2 + (size_t)l * HH * HH;
  bf16_t* d1 = w1T + (size_t)l * 64 * 160;
  bf16_t* d2 = w2T + (size_t)l * 64 * 64;
  for (int i = tid; i < 64 * 160; i += 256) {  // [n][k], K padded 144->160
    int n = i / 160, k = i % 160;
    d1[i] = (bf16_t)((k < 2 * HH + ED) ? w1[k * HH + n] : 0.0f);
  }
  for (int i = tid; i < 64 * 64; i += 256) {
    int n = i >> 6, k = i & 63;
    d2[i] = (bf16_t)w2[k * HH + n];
  }
}

// ---------------- node embedding: h = x @ emb_w + emb_b (bf16 out) -----------------
__global__ __launch_bounds__(256)
void k_embed(const float* __restrict__ x, const float* __restrict__ w,
             const float* __restrict__ bias, bf16_t* __restrict__ h) {
  __shared__ bf16_t sA[128 * 128];
  __shared__ bf16_t sBT[64 * 128];
  const int tid = threadIdx.x;
  const int n0  = blockIdx.x * 128;
  for (int i4 = tid; i4 < 128 * 32; i4 += 256) {
    int r = i4 >> 5, k = (i4 & 31) * 4;
    int ni = n0 + r;
    float4 v = make_float4(0.f, 0.f, 0.f, 0.f);
    if (ni < NN) v = *(const float4*)(x + (size_t)ni * ND + k);
    bf16_t* d = sA + r * 128 + k;
    d[0] = (bf16_t)v.x; d[1] = (bf16_t)v.y; d[2] = (bf16_t)v.z; d[3] = (bf16_t)v.w;
  }
  for (int i = tid; i < 64 * 128; i += 256) {
    int n = i >> 7, k = i & 127;
    sBT[i] = (bf16_t)w[k * HH + n];
  }
  __syncthreads();
  const int wave = tid >> 5, lane = tid & 31;
  v8f acc[4];
#pragma unroll
  for (int t = 0; t < 4; ++t) acc[t] = zero8();
#pragma unroll
  for (int kt = 0; kt < 4; ++kt) {
    v16bf a = ld_afrag(sA, wave * 16, 128, kt * 32, lane);
#pragma unroll
    for (int t = 0; t < 4; ++t)
      acc[t] = wmma_bf16(a, ld_bfragT(sBT, t * 16, 128, kt * 32, lane), acc[t]);
  }
  const int hi8 = (lane >> 4) << 3;
  const int cn  = lane & 15;
#pragma unroll
  for (int t = 0; t < 4; ++t)
#pragma unroll
    for (int j = 0; j < 8; ++j) {
      int r = wave * 16 + j + hi8;
      int ni = n0 + r;
      int c = t * 16 + cn;
      if (ni < NN) h[(size_t)ni * HH + c] = (bf16_t)(acc[t][j] + bias[c]);
    }
}

// -------- edge message MLP + scatter-add; gather + weights staged via async->LDS --------
__global__ __launch_bounds__(256)
void k_edge(const bf16_t* __restrict__ h, const int* __restrict__ eidx,
            const bf16_t* __restrict__ eattr_bf,
            const bf16_t* __restrict__ w1T, const float* __restrict__ b1,
            const bf16_t* __restrict__ w2T, const float* __restrict__ b2,
            float* __restrict__ agg) {
  __shared__ bf16_t sA[64 * 160];    // 20 KB, K padded 144 -> 160 (stride 320 B)
  __shared__ bf16_t sW1T[64 * 160];  // 20 KB
  __shared__ bf16_t sHid[64 * 64];   //  8 KB
  __shared__ bf16_t sW2T[64 * 64];   //  8 KB
  const int tid = threadIdx.x;
  const int e0  = blockIdx.x * 64;
  const unsigned baseA  = lds_off(sA);
  const unsigned baseW1 = lds_off(sW1T);
  const unsigned baseW2 = lds_off(sW2T);

  // node-feature gather: 4 parts x 64 edges, 64 B (32 bf16 cols) per part, pure async copy
  {
    int r = tid & 63, part = tid >> 6;          // parts 0,1: h[row]; 2,3: h[col]
    int e = e0 + r;
    int src = (part < 2) ? eidx[e] : eidx[EE + e];
    const bf16_t* gp = h + (size_t)src * HH + (part & 1) * 32;
    unsigned dl = baseA + (unsigned)(r * 320 + part * 64);
#pragma unroll
    for (int q = 0; q < 4; ++q) async_b128(dl + q * 16, (const char*)gp + q * 16);
  }
  // edge_attr: 32 B per edge, contiguous across edges
  if (tid < 128) {
    int r = tid >> 1, half = tid & 1;
    const bf16_t* gp = eattr_bf + (size_t)(e0 + r) * ED + half * 8;
    async_b128(baseA + (unsigned)(r * 320 + 256 + half * 16), gp);
  }
  // zero pad cols 144..159 (32 B per row) via LDS stores
  if (tid < 64) {
    uint4 z = make_uint4(0, 0, 0, 0);
    *(uint4*)(sA + tid * 160 + 144) = z;
    *(uint4*)(sA + tid * 160 + 152) = z;
  }
  // prestaged bf16 weights: straight block copies
  for (int c = tid; c < (64 * 160 * 2) / 16; c += 256)
    async_b128(baseW1 + c * 16, (const char*)w1T + c * 16);
  for (int c = tid; c < (64 * 64 * 2) / 16; c += 256)
    async_b128(baseW2 + c * 16, (const char*)w2T + c * 16);

  async_wait0();
  __syncthreads();

  const int wave = tid >> 5, lane = tid & 31;
  const int m   = wave >> 1;
  const int nt0 = (wave & 1) * 2;
  const int hi8 = (lane >> 4) << 3;
  const int cn  = lane & 15;

  v8f acc0 = zero8(), acc1 = zero8();
#pragma unroll
  for (int kt = 0; kt < 5; ++kt) {
    v16bf a = ld_afrag(sA, m * 16, 160, kt * 32, lane);
    acc0 = wmma_bf16(a, ld_bfragT(sW1T, nt0 * 16,      160, kt * 32, lane), acc0);
    acc1 = wmma_bf16(a, ld_bfragT(sW1T, nt0 * 16 + 16, 160, kt * 32, lane), acc1);
  }
#pragma unroll
  for (int j = 0; j < 8; ++j) {                 // bias + relu -> bf16 hidden
    int r  = m * 16 + j + hi8;
    int c0 = nt0 * 16 + cn, c1 = c0 + 16;
    float v0 = acc0[j] + b1[c0]; v0 = v0 > 0.f ? v0 : 0.f;
    float v1 = acc1[j] + b1[c1]; v1 = v1 > 0.f ? v1 : 0.f;
    sHid[r * 64 + c0] = (bf16_t)v0;
    sHid[r * 64 + c1] = (bf16_t)v1;
  }
  __syncthreads();
  acc0 = zero8(); acc1 = zero8();
#pragma unroll
  for (int kt = 0; kt < 2; ++kt) {
    v16bf a = ld_afrag(sHid, m * 16, 64, kt * 32, lane);
    acc0 = wmma_bf16(a, ld_bfragT(sW2T, nt0 * 16,      64, kt * 32, lane), acc0);
    acc1 = wmma_bf16(a, ld_bfragT(sW2T, nt0 * 16 + 16, 64, kt * 32, lane), acc1);
  }
#pragma unroll
  for (int j = 0; j < 8; ++j) {                 // scatter-add into agg[row]
    int r = m * 16 + j + hi8;
    int e = e0 + r;
    int dst = eidx[e];
    int c0 = nt0 * 16 + cn, c1 = c0 + 16;
    atomicAdd(&agg[(size_t)dst * HH + c0], acc0[j] + b2[c0]);
    atomicAdd(&agg[(size_t)dst * HH + c1], acc1[j] + b2[c1]);
  }
}

// ---- GRU cell: gsum = [agg|h] @ [wih^T;whh^T] (K=128), ghn = h @ whh_n^T, gates fused ----
__global__ __launch_bounds__(256)
void k_gru(const float* __restrict__ agg, const bf16_t* __restrict__ hin,
           const float* __restrict__ wih, const float* __restrict__ whh,
           const float* __restrict__ bih, const float* __restrict__ bhh,
           bf16_t* __restrict__ hout) {
  __shared__ bf16_t sAcat[32 * 128];  //  8 KB  [node][k]: k<64 agg, k>=64 h
  __shared__ float  sGsum[32 * 192];  // 24 KB
  __shared__ float  sGhn [32 * 64];   //  8 KB
  const int tid = threadIdx.x;
  const int n0  = blockIdx.x * 32;
  for (int i = tid; i < 32 * 128; i += 256) {
    int r = i >> 7, k = i & 127;
    int ni = n0 + r;
    bf16_t v = (bf16_t)0.0f;
    if (ni < NN) v = (k < 64) ? (bf16_t)agg[(size_t)ni * HH + k]
                              : hin[(size_t)ni * HH + (k - 64)];
    sAcat[i] = v;
  }
  __syncthreads();
  const int wave = tid >> 5, lane = tid & 31;
  const int hi8  = (lane >> 4) << 3;
  const int cn   = lane & 15;
  const int kb16 = (lane >> 4) << 4;
#pragma unroll
  for (int i = 0; i < 4; ++i) {
    int t = wave * 4 + i;                 // 0..23: gsum (2M x 12N); 24..31: ghn (2M x 4N)
    v8f acc = zero8();
    if (t < 24) {
      int m = t / 12, n = t % 12;
      int ng = n * 16 + cn;
#pragma unroll
      for (int kt = 0; kt < 4; ++kt) {
        v16bf a = ld_afrag(sAcat, m * 16, 128, kt * 32, lane);
        int kk = kt * 32 + kb16;
        const float* bp = (kk < 64) ? (wih + (size_t)ng * HH + kk)
                                    : (whh + (size_t)ng * HH + (kk - 64));
        acc = wmma_bf16(a, ld_bfrag_g(bp), acc);
      }
#pragma unroll
      for (int j = 0; j < 8; ++j) sGsum[(m * 16 + j + hi8) * 192 + ng] = acc[j];
    } else {
      int tt = t - 24;
      int m = tt >> 2, n = tt & 3;
      int ng = n * 16 + cn;
#pragma unroll
      for (int kt = 0; kt < 2; ++kt) {
        v16bf a = ld_afrag(sAcat, m * 16, 128, 64 + kt * 32, lane);
        int kk = kt * 32 + kb16;
        const float* bp = whh + (size_t)(128 + ng) * HH + kk;
        acc = wmma_bf16(a, ld_bfrag_g(bp), acc);
      }
#pragma unroll
      for (int j = 0; j < 8; ++j) sGhn[(m * 16 + j + hi8) * 64 + ng] = acc[j];
    }
  }
  __syncthreads();
  for (int i = tid; i < 32 * 64; i += 256) {
    int r = i >> 6, c = i & 63;
    int ni = n0 + r;
    if (ni >= NN) continue;
    float gr = sGsum[r * 192 + c]      + bih[c]      + bhh[c];
    float gz = sGsum[r * 192 + 64 + c] + bih[64 + c] + bhh[64 + c];
    float gs_n    = sGsum[r * 192 + 128 + c];
    float ghn_raw = sGhn[r * 64 + c];
    float gin = gs_n - ghn_raw + bih[128 + c];
    float ghn = ghn_raw + bhh[128 + c];
    float rg = 1.0f / (1.0f + __expf(-gr));
    float zg = 1.0f / (1.0f + __expf(-gz));
    float ng = tanhf(gin + rg * ghn);
    float ho = (float)hin[(size_t)ni * HH + c];
    hout[(size_t)ni * HH + c] = (bf16_t)((1.0f - zg) * ng + zg * ho);
  }
}

// ---------------- pooling + readout ----------------
__global__ __launch_bounds__(256)
void k_pool(const bf16_t* __restrict__ h, const int* __restrict__ batch,
            float* __restrict__ sums, float* __restrict__ cnts) {
  int i = blockIdx.x * 256 + threadIdx.x;
  if (i >= NN * HH) return;
  int ni = i >> 6, c = i & 63;
  int g = batch[ni];
  atomicAdd(&sums[g * HH + c], (float)h[i]);
  if (c == 0) atomicAdd(&cnts[g], 1.0f);
}

__global__ __launch_bounds__(256)
void k_readout(const float* __restrict__ sums, const float* __restrict__ cnts,
               const float* __restrict__ w1, const float* __restrict__ b1v,
               const float* __restrict__ w2, const float* __restrict__ b2v,
               float* __restrict__ out) {
  __shared__ float gmean[GG * HH];
  __shared__ float hid[GG * HH];
  int tid = threadIdx.x;
  for (int i = tid; i < GG * HH; i += 256) {
    int g = i >> 6;
    gmean[i] = sums[i] / fmaxf(cnts[g], 1.0f);
  }
  __syncthreads();
  for (int i = tid; i < GG * HH; i += 256) {
    int g = i >> 6, j = i & 63;
    float s = b1v[j];
    for (int k = 0; k < HH; ++k) s += gmean[g * HH + k] * w1[k * HH + j];
    hid[i] = s > 0.f ? s : 0.f;
  }
  __syncthreads();
  if (tid < GG) {
    float s = b2v[0];
    for (int j = 0; j < HH; ++j) s += hid[tid * HH + j] * w2[j];
    out[tid] = 1.0f / (1.0f + __expf(-s));
  }
}

__global__ void k_zero(float* p, int n) {
  int i = blockIdx.x * 256 + threadIdx.x;
  if (i < n) p[i] = 0.0f;
}

extern "C" void kernel_launch(void* const* d_in, const int* in_sizes, int n_in,
                              void* d_out, int out_size, void* d_ws, size_t ws_size,
                              hipStream_t stream) {
  (void)in_sizes; (void)n_in; (void)out_size; (void)ws_size;
  const float* x       = (const float*)d_in[0];
  const int*   eidx    = (const int*)d_in[1];
  const float* eattr   = (const float*)d_in[2];
  const int*   batch   = (const int*)d_in[3];
  const float* emb_w   = (const float*)d_in[4];
  const float* emb_b   = (const float*)d_in[5];
  const float* msg_w1  = (const float*)d_in[6];
  const float* msg_b1  = (const float*)d_in[7];
  const float* msg_w2  = (const float*)d_in[8];
  const float* msg_b2  = (const float*)d_in[9];
  const float* gru_wih = (const float*)d_in[10];
  const float* gru_whh = (const float*)d_in[11];
  const float* gru_bih = (const float*)d_in[12];
  const float* gru_bhh = (const float*)d_in[13];
  const float* ro_w1   = (const float*)d_in[14];
  const float* ro_b1   = (const float*)d_in[15];
  const float* ro_w2   = (const float*)d_in[16];
  const float* ro_b2   = (const float*)d_in[17];

  char* ws = (char*)d_ws;
  size_t off = 0;
  bf16_t* hA      = (bf16_t*)(ws + off); off += (size_t)NN * HH * 2;   // 6.4 MB
  bf16_t* hB      = (bf16_t*)(ws + off); off += (size_t)NN * HH * 2;   // 6.4 MB
  float*  agg     = (float*) (ws + off); off += (size_t)NN * HH * 4;   // 12.8 MB
  bf16_t* eattrbf = (bf16_t*)(ws + off); off += (size_t)EE * ED * 2;   // 25.6 MB
  bf16_t* w1T     = (bf16_t*)(ws + off); off += (size_t)LL * 64 * 160 * 2;
  bf16_t* w2T     = (bf16_t*)(ws + off); off += (size_t)LL * 64 * 64 * 2;
  float*  psum    = (float*) (ws + off); off += (size_t)GG * HH * 4;
  float*  pcnt    = (float*) (ws + off);

  k_prep_attr<<<(EE * ED / 4 + 255) / 256, 256, 0, stream>>>(eattr, eattrbf);
  k_prep_w<<<LL, 256, 0, stream>>>(msg_w1, msg_w2, w1T, w2T);
  k_embed<<<(NN + 127) / 128, 256, 0, stream>>>(x, emb_w, emb_b, hA);

  bf16_t* hc = hA;
  bf16_t* hn = hB;
  for (int l = 0; l < LL; ++l) {
    k_zero<<<(NN * HH + 255) / 256, 256, 0, stream>>>(agg, NN * HH);
    k_edge<<<EE / 64, 256, 0, stream>>>(hc, eidx, eattrbf,
        w1T + (size_t)l * 64 * 160, msg_b1 + l * HH,
        w2T + (size_t)l * 64 * 64,  msg_b2 + l * HH, agg);
    k_gru<<<(NN + 31) / 32, 256, 0, stream>>>(agg, hc,
        gru_wih + (size_t)l * 3 * HH * HH, gru_whh + (size_t)l * 3 * HH * HH,
        gru_bih + l * 3 * HH, gru_bhh + l * 3 * HH, hn);
    bf16_t* t = hc; hc = hn; hn = t;
  }
  k_zero<<<(GG * HH + GG + 255) / 256, 256, 0, stream>>>(psum, GG * HH + GG);
  k_pool<<<(NN * HH + 255) / 256, 256, 0, stream>>>(hc, batch, psum, pcnt);
  k_readout<<<1, 256, 0, stream>>>(psum, pcnt, ro_w1, ro_b1, ro_w2, ro_b2, (float*)d_out);
}
